// MaxFlowModel_63333587747292
// MI455X (gfx1250) — matrix operand order repaired and checked by reading, not compiled, not run on hardware
//
#include <hip/hip_runtime.h>
#include <hip/hip_bf16.h>
#include <math.h>

typedef __attribute__((ext_vector_type(16))) _Float16 v16h;
typedef __attribute__((ext_vector_type(8)))  float    v8f;

constexpr int kB   = 4;
constexpr int kN   = 1024;
constexpr int kD   = 32;
constexpr int kH   = 4;
constexpr int kFin = 16;
constexpr int kL   = 2;

// ---------------------------------------------------------------------------
// WMMA helpers (wave32, 16x16x32 f16 -> f32)
// A operand: LDS row-major [16][lda] halves.
//   lane L holds M = L&15; VGPR layout: lanes 0-15 K in {0..15},
//   lanes 16-31 K in {16..31} (same contiguous K order as B).
// B operand: transposed buffer Bt[n][k] (row per output column) so the
//   per-lane gather (N = lane&15 (+nb), K = k0 + (lane>=16?16:0) + 0..15)
//   is 16 contiguous halves (two b128 loads).
// ---------------------------------------------------------------------------
__device__ __forceinline__ v16h ldA16(const _Float16* A, int lda, int lane, int k0) {
  const int m  = lane & 15;
  const int kb = (lane >> 4) << 4;
  const _Float16* p = A + m * lda + k0 + kb;
  v16h a;
#pragma unroll
  for (int e = 0; e < 8; ++e) { a[e] = p[e]; a[8 + e] = p[8 + e]; }
  return a;
}

__device__ __forceinline__ v16h ldBt16(const _Float16* Bt, int ldb, int lane, int nb, int k0) {
  const int n  = (lane & 15) + nb;
  const int kb = (lane >> 4) << 4;
  const _Float16* p = Bt + n * ldb + k0 + kb;
  v16h b;
#pragma unroll
  for (int e = 0; e < 16; ++e) b[e] = p[e];
  return b;
}

__device__ __forceinline__ v8f wmma16(v16h a, v16h b, v8f c) {
  return __builtin_amdgcn_wmma_f32_16x16x32_f16(false, a, false, b, (short)0, c,
                                                false, false);
}

// ---------------------------------------------------------------------------
// 1) node_enc = tanh(node_input @ W_node_enc + b)   [B,N,16]@[16,32]
// ---------------------------------------------------------------------------
__global__ void k_node_enc(const float* __restrict__ x, const float* __restrict__ W,
                           const float* __restrict__ bias, float* __restrict__ out) {
  int i = blockIdx.x * blockDim.x + threadIdx.x;
  if (i >= kB * kN * kD) return;
  int d   = i & (kD - 1);
  int row = i >> 5;
  float s = bias[d];
#pragma unroll
  for (int f = 0; f < kFin; ++f) s += x[row * kFin + f] * W[f * kD + d];
  out[i] = tanhf(s);
}

// ---------------------------------------------------------------------------
// 2) W_edge_enc [N,D] f32 -> Wt [D][N] f16 (transposed for B-operand gather)
// ---------------------------------------------------------------------------
__global__ void k_wt_f16(const float* __restrict__ W, _Float16* __restrict__ Wt) {
  int i = blockIdx.x * blockDim.x + threadIdx.x;
  if (i >= kN * kD) return;
  int d = i & (kD - 1);
  int n = i >> 5;
  Wt[d * kN + n] = (_Float16)W[n * kD + d];
}

// ---------------------------------------------------------------------------
// 3) edge_enc = tanh(edge_input[4096,1024] @ W[1024,32] + b)  (WMMA GEMM)
//    one wave per 16x32 output tile, K in chunks of 32
// ---------------------------------------------------------------------------
__global__ void __launch_bounds__(32)
k_edge_enc(const float* __restrict__ E, const _Float16* __restrict__ Wt,
           const float* __restrict__ bias, float* __restrict__ out) {
  __shared__ _Float16 As[16 * 32];
  const int lane = threadIdx.x;
  const int row0 = blockIdx.x * 16;
  v8f c0 = {}, c1 = {};
  for (int kt = 0; kt < kN / 32; ++kt) {
    const int r = lane >> 1, cb = (lane & 1) * 16;
    const float* src = E + (size_t)(row0 + r) * kN + kt * 32 + cb;
#pragma unroll
    for (int e = 0; e < 16; ++e) As[r * 32 + cb + e] = (_Float16)src[e];
    __syncthreads();
    v16h a  = ldA16(As, 32, lane, 0);
    v16h b0 = ldBt16(Wt, kN, lane, 0,  kt * 32);
    v16h b1 = ldBt16(Wt, kN, lane, 16, kt * 32);
    c0 = wmma16(a, b0, c0);
    c1 = wmma16(a, b1, c1);
    __syncthreads();
  }
  const int d0 = lane & 15, hi = (lane >> 4) * 8;
#pragma unroll
  for (int r = 0; r < 8; ++r) {
    int row = row0 + r + hi;
    out[row * kD + d0]      = tanhf(c0[r] + bias[d0]);
    out[row * kD + d0 + 16] = tanhf(c1[r] + bias[d0 + 16]);
  }
}

// ---------------------------------------------------------------------------
// 4) per (b,head,16-row tile): h = x@Wh (WMMA), f1 = h.a1, f2 = h.a2,
//    store h transposed f16 [D][N] for the aggregation B-operand.
// ---------------------------------------------------------------------------
__global__ void __launch_bounds__(32)
k_gat_h(const float* __restrict__ x,   // [B,N,D]
        const float* __restrict__ Wh,  // [H,D,D] (layer slice)
        const float* __restrict__ a1,  // [H,D]
        const float* __restrict__ a2,  // [H,D]
        _Float16* __restrict__ hT,     // [B,H,D,N]
        float* __restrict__ f1, float* __restrict__ f2) {
  __shared__ _Float16 WhT[kD * kD];
  __shared__ _Float16 As[16 * kD];
  __shared__ float    hS[16 * kD];
  const int lane = threadIdx.x;
  const int tile = blockIdx.x & 63;
  const int head = (blockIdx.x >> 6) & 3;
  const int b    = blockIdx.x >> 8;
  const int n0   = tile * 16;
  const int bh   = b * kH + head;

  const float* Wsrc = Wh + head * kD * kD;
#pragma unroll
  for (int f = 0; f < kD; ++f) WhT[lane * kD + f] = (_Float16)Wsrc[f * kD + lane];
  {
    const int r = lane >> 1, cb = (lane & 1) * 16;
    const float* src = x + ((size_t)b * kN + n0 + r) * kD + cb;
#pragma unroll
    for (int e = 0; e < 16; ++e) As[r * kD + cb + e] = (_Float16)src[e];
  }
  __syncthreads();

  v8f c0 = {}, c1 = {};
  v16h a  = ldA16(As, kD, lane, 0);
  v16h b0 = ldBt16(WhT, kD, lane, 0,  0);
  v16h b1 = ldBt16(WhT, kD, lane, 16, 0);
  c0 = wmma16(a, b0, c0);
  c1 = wmma16(a, b1, c1);

  const int d0 = lane & 15, hi = (lane >> 4) * 8;
#pragma unroll
  for (int r = 0; r < 8; ++r) {
    hS[(r + hi) * kD + d0]      = c0[r];
    hS[(r + hi) * kD + d0 + 16] = c1[r];
  }
  __syncthreads();

  if (lane < 16) {
    float s = 0.f;
#pragma unroll
    for (int d = 0; d < kD; ++d) s += hS[lane * kD + d] * a1[head * kD + d];
    f1[bh * kN + n0 + lane] = s;
  } else {
    int r = lane - 16;
    float s = 0.f;
#pragma unroll
    for (int d = 0; d < kD; ++d) s += hS[r * kD + d] * a2[head * kD + d];
    f2[bh * kN + n0 + r] = s;
  }

  __align__(32) _Float16 buf[16];
#pragma unroll
  for (int i = 0; i < 16; ++i) buf[i] = (_Float16)hS[i * kD + lane];
  *(v16h*)(hT + ((size_t)bh * kD + lane) * kN + n0) = *(const v16h*)buf;
}

// ---------------------------------------------------------------------------
// 5) aggregation: softmax(leakyrelu(f1+f2)+bias) @ h, per (b,head,16-row tile)
//    Unnormalized exp stored as f16 alpha in LDS (32KB); 1/rowsum folded into
//    the epilogue (softmax normalization is linear in the output row).
// ---------------------------------------------------------------------------
__global__ void __launch_bounds__(32)
k_gat_agg(const float* __restrict__ f1, const float* __restrict__ f2,
          const float* __restrict__ bias,   // [B,N,N]
          const _Float16* __restrict__ hT,  // [B,H,D,N]
          float* __restrict__ outh) {       // [B,H,N,D]
  __shared__ _Float16 alpha[16 * kN];  // 32 KB
  __shared__ float    f2s[kN];
  __shared__ float    rowbuf[kN];
  __shared__ float    f1s[16];
  __shared__ float    rscale[16];
  const int lane = threadIdx.x;
  const int tile = blockIdx.x & 63;
  const int head = (blockIdx.x >> 6) & 3;
  const int b    = blockIdx.x >> 8;
  const int n0   = tile * 16;
  const int bh   = b * kH + head;

  for (int m = lane; m < kN; m += 32) f2s[m] = f2[bh * kN + m];
  if (lane < 16) f1s[lane] = f1[bh * kN + n0 + lane];
  __syncthreads();

  for (int r = 0; r < 16; ++r) {
    const float fr = f1s[r];
    const float* brow = bias + ((size_t)(b * kN + n0 + r)) * kN;
    if (r < 15)  // prefetch next bias row
      __builtin_prefetch(bias + ((size_t)(b * kN + n0 + r + 1)) * kN + lane * 4, 0, 3);
    float mx = -3.4e38f;
    for (int m = lane; m < kN; m += 32) {
      float l = fr + f2s[m];
      l = (l > 0.f) ? l : 0.2f * l;
      l += brow[m];
      rowbuf[m] = l;
      mx = fmaxf(mx, l);
    }
#pragma unroll
    for (int off = 16; off > 0; off >>= 1) mx = fmaxf(mx, __shfl_xor(mx, off));
    float sum = 0.f;
    for (int m = lane; m < kN; m += 32) {
      float e = __expf(rowbuf[m] - mx);
      alpha[r * kN + m] = (_Float16)e;
      sum += e;
    }
#pragma unroll
    for (int off = 16; off > 0; off >>= 1) sum += __shfl_xor(sum, off);
    if (lane == 0) rscale[r] = 1.f / sum;
    __syncthreads();
  }

  v8f c0 = {}, c1 = {};
  const _Float16* hTb = hT + (size_t)bh * kD * kN;
  for (int kt = 0; kt < kN / 32; ++kt) {
    v16h a  = ldA16(alpha, kN, lane, kt * 32);
    v16h b0 = ldBt16(hTb, kN, lane, 0,  kt * 32);
    v16h b1 = ldBt16(hTb, kN, lane, 16, kt * 32);
    c0 = wmma16(a, b0, c0);
    c1 = wmma16(a, b1, c1);
  }

  const int d0 = lane & 15, hi = (lane >> 4) * 8;
#pragma unroll
  for (int r = 0; r < 8; ++r) {
    int rr = r + hi;
    float s = rscale[rr];
    float* dst = outh + ((size_t)bh * kN + n0 + rr) * kD;
    dst[d0]      = c0[r] * s;
    dst[d0 + 16] = c1[r] * s;
  }
}

// ---------------------------------------------------------------------------
// 6) gate: cur = tanh(mean_head gat), z = sigmoid(prev@W1 + cur@W2 + b),
//    out = z*prev + (1-z)*cur.  8 rows per 256-thread block.
// ---------------------------------------------------------------------------
__global__ void __launch_bounds__(256)
k_gate(const float* __restrict__ prev, const float* __restrict__ gath, // [B,H,N,D]
       const float* __restrict__ W1, const float* __restrict__ W2,
       const float* __restrict__ bvec, float* __restrict__ outp) {
  __shared__ float W1s[kD * kD], W2s[kD * kD];
  __shared__ float curS[8 * kD], prevS[8 * kD];
  const int t  = threadIdx.x;
  const int rl = t >> 5, d = t & 31;
  const int row = blockIdx.x * 8 + rl;       // over B*N
  for (int i = t; i < kD * kD; i += 256) { W1s[i] = W1[i]; W2s[i] = W2[i]; }
  const int b = row >> 10;
  const int n = row & 1023;
  size_t base = ((size_t)(b * kH) * kN + n) * kD + d;
  const size_t hstride = (size_t)kN * kD;
  float m = gath[base] + gath[base + hstride] + gath[base + 2 * hstride] +
            gath[base + 3 * hstride];
  float cur = tanhf(0.25f * m);
  float pv  = prev[(size_t)row * kD + d];
  curS[rl * kD + d]  = cur;
  prevS[rl * kD + d] = pv;
  __syncthreads();
  float z = bvec[d];
#pragma unroll
  for (int k = 0; k < kD; ++k)
    z += prevS[rl * kD + k] * W1s[k * kD + d] + curS[rl * kD + k] * W2s[k * kD + d];
  z = 1.f / (1.f + __expf(-z));
  outp[(size_t)row * kD + d] = z * pv + (1.f - z) * cur;
}

// ---------------------------------------------------------------------------
// 7) decode: ew = sigmoid(edge_enc @ W_dec + b_dec)
// ---------------------------------------------------------------------------
__global__ void k_decode(const float* __restrict__ enc, const float* __restrict__ Wd,
                         const float* __restrict__ bd, float* __restrict__ ew) {
  int row = blockIdx.x * blockDim.x + threadIdx.x;
  if (row >= kB * kN) return;
  float s = bd[0];
#pragma unroll
  for (int d = 0; d < kD; ++d) s += enc[row * kD + d] * Wd[d];
  ew[row] = 1.f / (1.f + __expf(-s));
}

// ---------------------------------------------------------------------------
// 8) flow = edge_input * ew (broadcast over last dim); deterministic partial
//    sums of mask*flow per block, reduced in k_reduce (no float atomics).
// ---------------------------------------------------------------------------
__global__ void __launch_bounds__(256)
k_flow(const float* __restrict__ E, const float* __restrict__ mask,
       const float* __restrict__ ew, float* __restrict__ flow_out,
       float* __restrict__ partials) {
  __shared__ float red[256];
  size_t i = (size_t)blockIdx.x * 256 + threadIdx.x;
  int b = (int)(i >> 20);
  int n = (int)((i >> 10) & 1023);
  float v = E[i] * ew[b * kN + n];
  flow_out[i] = v;
  red[threadIdx.x] = mask[i] * v;
  __syncthreads();
  for (int s = 128; s > 0; s >>= 1) {
    if (threadIdx.x < s) red[threadIdx.x] += red[threadIdx.x + s];
    __syncthreads();
  }
  if (threadIdx.x == 0) partials[blockIdx.x] = red[0];
}

__global__ void __launch_bounds__(256)
k_reduce(const float* __restrict__ partials, float* __restrict__ out) {
  __shared__ float red[256];
  int b = blockIdx.x;
  float s = 0.f;
  for (int i = threadIdx.x; i < 4096; i += 256) s += partials[b * 4096 + i];
  red[threadIdx.x] = s;
  __syncthreads();
  for (int st = 128; st > 0; st >>= 1) {
    if (threadIdx.x < st) red[threadIdx.x] += red[threadIdx.x + st];
    __syncthreads();
  }
  if (threadIdx.x == 0) out[b] = red[0];
}

// ---------------------------------------------------------------------------
extern "C" void kernel_launch(void* const* d_in, const int* in_sizes, int n_in,
                              void* d_out, int out_size, void* d_ws, size_t ws_size,
                              hipStream_t stream) {
  const float* node_input  = (const float*)d_in[0];
  const float* edge_input  = (const float*)d_in[1];
  const float* node_bias   = (const float*)d_in[2];
  const float* edge_bias   = (const float*)d_in[3];
  const float* source_mask = (const float*)d_in[4];
  const float* W_node_enc  = (const float*)d_in[5];
  const float* b_node_enc  = (const float*)d_in[6];
  const float* W_edge_enc  = (const float*)d_in[7];
  const float* b_edge_enc  = (const float*)d_in[8];
  const float* Wn_gat = (const float*)d_in[9];
  const float* an1    = (const float*)d_in[10];
  const float* an2    = (const float*)d_in[11];
  const float* We_gat = (const float*)d_in[12];
  const float* ae1    = (const float*)d_in[13];
  const float* ae2    = (const float*)d_in[14];
  const float* Wng1   = (const float*)d_in[15];
  const float* Wng2   = (const float*)d_in[16];
  const float* bng    = (const float*)d_in[17];
  const float* Weg1   = (const float*)d_in[18];
  const float* Weg2   = (const float*)d_in[19];
  const float* beg    = (const float*)d_in[20];
  const float* W_dec  = (const float*)d_in[21];
  const float* b_dec  = (const float*)d_in[22];
  (void)in_sizes; (void)n_in; (void)out_size; (void)ws_size;

  char* ws = (char*)d_ws;
  size_t off = 0;
  auto alloc = [&](size_t bytes) -> char* {
    char* p = ws + off;
    off += (bytes + 255) & ~(size_t)255;
    return p;
  };
  const size_t encElems = (size_t)kB * kN * kD;
  float*    node0 = (float*)alloc(encElems * 4);
  float*    node1 = (float*)alloc(encElems * 4);
  float*    edge0 = (float*)alloc(encElems * 4);
  float*    edge1 = (float*)alloc(encElems * 4);
  _Float16* Wt    = (_Float16*)alloc((size_t)kN * kD * 2);
  _Float16* hT    = (_Float16*)alloc((size_t)kB * kH * kD * kN * 2);
  float*    f1    = (float*)alloc((size_t)kB * kH * kN * 4);
  float*    f2    = (float*)alloc((size_t)kB * kH * kN * 4);
  float*    gath  = (float*)alloc((size_t)kB * kH * kN * kD * 4);
  float*    ew    = (float*)alloc((size_t)kB * kN * 4);
  float*    part  = (float*)alloc((size_t)kB * 4096 * 4);

  // encoders
  k_node_enc<<<(kB * kN * kD + 255) / 256, 256, 0, stream>>>(node_input, W_node_enc,
                                                             b_node_enc, node0);
  k_wt_f16<<<(kN * kD + 255) / 256, 256, 0, stream>>>(W_edge_enc, Wt);
  k_edge_enc<<<kB * kN / 16, 32, 0, stream>>>(edge_input, Wt, b_edge_enc, edge0);

  float *ncur = node0, *nnxt = node1;
  float *ecur = edge0, *enxt = edge1;
  const int gatGrid = kB * kH * (kN / 16);

  for (int l = 0; l < kL; ++l) {
    // node GAT
    k_gat_h<<<gatGrid, 32, 0, stream>>>(ncur, Wn_gat + (size_t)l * kH * kD * kD,
                                        an1 + (size_t)l * kH * kD,
                                        an2 + (size_t)l * kH * kD, hT, f1, f2);
    k_gat_agg<<<gatGrid, 32, 0, stream>>>(f1, f2, node_bias, hT, gath);
    k_gate<<<kB * kN / 8, 256, 0, stream>>>(ncur, gath, Wng1 + (size_t)l * kD * kD,
                                            Wng2 + (size_t)l * kD * kD,
                                            bng + (size_t)l * kD, nnxt);
    // edge GAT
    k_gat_h<<<gatGrid, 32, 0, stream>>>(ecur, We_gat + (size_t)l * kH * kD * kD,
                                        ae1 + (size_t)l * kH * kD,
                                        ae2 + (size_t)l * kH * kD, hT, f1, f2);
    k_gat_agg<<<gatGrid, 32, 0, stream>>>(f1, f2, edge_bias, hT, gath);
    k_gate<<<kB * kN / 8, 256, 0, stream>>>(ecur, gath, Weg1 + (size_t)l * kD * kD,
                                            Weg2 + (size_t)l * kD * kD,
                                            beg + (size_t)l * kD, enxt);
    float* t;
    t = ncur; ncur = nnxt; nnxt = t;
    t = ecur; ecur = enxt; enxt = t;
  }

  k_decode<<<(kB * kN + 255) / 256, 256, 0, stream>>>(ecur, W_dec, b_dec, ew);

  float* out = (float*)d_out;  // [0..3] = max_flow, [4..] = flow_vals
  k_flow<<<kB * kN * kN / 256, 256, 0, stream>>>(edge_input, source_mask, ew,
                                                 out + 4, part);
  k_reduce<<<kB, 256, 0, stream>>>(part, out);
}
